// BiLSTM_CRF_7181185319212
// MI455X (gfx1250) — matrix operand order repaired and checked by reading, not compiled, not run on hardware
//
#include <hip/hip_runtime.h>
#include <hip/hip_bf16.h>

typedef __bf16 bf16;
typedef __attribute__((ext_vector_type(16))) __bf16 v16bf;
typedef __attribute__((ext_vector_type(8)))  __bf16 v8bf;
typedef __attribute__((ext_vector_type(8)))  float  v8f;

#define TT   512
#define BB   16
#define HH   256
#define G4H  1024
#define NTAG 6
#define START_TAG 4
#define STOP_TAG  5
#define NEGV -10000.0f
#define HSTRIDE 264   // 16-row LDS h buffer, padded to dodge bank conflicts

// ---------------- WMMA fragment loaders (bf16, 16x16x32) ----------------
// A (16x32, row-major src, stride in elements):
// lane l: row = l&15 ; khalf = (l>=16)?8:0 ; holds K = khalf+[0..8) and khalf+16+[0..8)
__device__ inline v16bf load_a_frag(const bf16* base, int stride, int lane) {
    int r = lane & 15;
    int khalf = (lane >> 4) << 3;
    const bf16* p = base + (size_t)r * stride + khalf;
    union { v16bf v; v8bf h[2]; } u;
    u.h[0] = *(const v8bf*)(p);
    u.h[1] = *(const v8bf*)(p + 16);
    return u.v;
}
// B (32x16): src row-major [N][K] (i.e. W with output rows), lane l: col = l&15,
// khalf = (l>=16)?8:0 ; holds K = khalf+[0..8) and khalf+16+[0..8)
__device__ inline v16bf load_b_frag(const bf16* base, int stride, int lane) {
    int n = lane & 15;
    int khalf = (lane >> 4) << 3;
    const bf16* p = base + (size_t)n * stride + khalf;
    union { v16bf v; v8bf h[2]; } u;
    u.h[0] = *(const v8bf*)(p);
    u.h[1] = *(const v8bf*)(p + 16);
    return u.v;
}

__device__ inline float fast_sigmoid(float x) { return 1.0f / (1.0f + __expf(-x)); }
__device__ inline float fast_tanh(float x) {
    float ax = fabsf(x);
    float e = __expf(-2.0f * ax);
    float t = (1.0f - e) / (1.0f + e);
    return copysignf(t, x);
}

// ------------- Phase 0: gather embeddings -> bf16, weights -> bf16, bias fold -------------
__global__ void prep_kernel(const int* __restrict__ sentence,
                            const float* __restrict__ emb,
                            const float* __restrict__ W_ih,
                            const float* __restrict__ W_hh,
                            const float* __restrict__ b_ih,
                            const float* __restrict__ b_hh,
                            bf16* __restrict__ E, bf16* __restrict__ Wihb,
                            bf16* __restrict__ Whhb, float* __restrict__ bias) {
    const int NE = TT * BB * HH;       // 2,097,152
    const int NW = G4H * HH;           // 262,144
    int idx = blockIdx.x * blockDim.x + threadIdx.x;
    if (idx < NE) {
        int k   = idx & (HH - 1);
        int row = idx >> 8;            // row = t*16 + b
        int b = row & 15, t = row >> 4;
        int tok = sentence[b * TT + t];
        E[idx] = (bf16)emb[(size_t)tok * HH + k];
    } else if (idx < NE + NW) {
        int i = idx - NE;        Wihb[i] = (bf16)W_ih[i];
    } else if (idx < NE + 2 * NW) {
        int i = idx - NE - NW;   Whhb[i] = (bf16)W_hh[i];
    } else if (idx < NE + 2 * NW + G4H) {
        int i = idx - NE - 2 * NW; bias[i] = b_ih[i] + b_hh[i];
    }
}

// ------------- Phase 1: X[t*16+b, 0:1024] = E @ W_ih^T + (b_ih+b_hh), parallel WMMA -------------
__global__ __launch_bounds__(256) void gemm_x_kernel(const bf16* __restrict__ E,
                                                     const bf16* __restrict__ Wih,
                                                     const float* __restrict__ bias,
                                                     float* __restrict__ X) {
    int lane = threadIdx.x & 31;
    int wave = threadIdx.x >> 5;
    int tile = blockIdx.x * 8 + wave;   // 32768 tiles total
    int tm = tile >> 6;                 // 0..511 (row tile = time step, rows t*16+b)
    int tn = tile & 63;                 // 0..63  (16-col slice of 4H)
    int n0 = tn * 16;
    int n   = lane & 15;
    int mlo = (lane >> 4) << 3;
    float binit = bias[n0 + n];
    v8f acc;
#pragma unroll
    for (int j = 0; j < 8; ++j) acc[j] = binit;
#pragma unroll
    for (int kk = 0; kk < 8; ++kk) {
        v16bf a = load_a_frag(E + (size_t)tm * 16 * HH + kk * 32, HH, lane);
        v16bf b = load_b_frag(Wih + (size_t)n0 * HH + kk * 32, HH, lane);
        acc = __builtin_amdgcn_wmma_f32_16x16x32_bf16(false, a, false, b,
                                                      (short)0, acc, false, false);
    }
#pragma unroll
    for (int j = 0; j < 8; ++j)
        X[(size_t)(tm * 16 + mlo + j) * G4H + n0 + n] = acc[j];
}

// ------------- Phase 2: serial LSTM recurrence, single workgroup, c in regs, h in LDS -------------
__global__ __launch_bounds__(512) void lstm_kernel(const float* __restrict__ X,
                                                   const bf16* __restrict__ Whh,
                                                   bf16* __restrict__ HS) {
    __shared__ bf16 h_lds[16 * HSTRIDE];
    int tid  = threadIdx.x;
    int lane = tid & 31;
    int hc   = tid >> 5;                // wave id == 16-col slice of H
    for (int i = tid; i < 16 * HSTRIDE; i += 512) h_lds[i] = (bf16)0.0f;
    __syncthreads();

    const int mlo = (lane >> 4) << 3;   // batch-row base of this half-wave
    const int n   = lane & 15;
    v8f c;
#pragma unroll
    for (int j = 0; j < 8; ++j) c[j] = 0.0f;

    for (int t = 0; t < TT; ++t) {
        const float* Xr = X + (size_t)t * 16 * G4H;
        v8f acc[4];
#pragma unroll
        for (int g = 0; g < 4; ++g) {
            int n0 = g * 256 + hc * 16;
#pragma unroll
            for (int j = 0; j < 8; ++j)
                acc[g][j] = Xr[(size_t)(mlo + j) * G4H + n0 + n];
        }
#pragma unroll
        for (int kk = 0; kk < 8; ++kk) {
            v16bf a = load_a_frag(h_lds + kk * 32, HSTRIDE, lane);
#pragma unroll
            for (int g = 0; g < 4; ++g) {
                v16bf b = load_b_frag(Whh + (size_t)(g * 256 + hc * 16) * HH + kk * 32,
                                      HH, lane);
                acc[g] = __builtin_amdgcn_wmma_f32_16x16x32_bf16(false, a, false, b,
                                                                 (short)0, acc[g],
                                                                 false, false);
            }
        }
        __syncthreads();   // all waves done reading h(t-1)
#pragma unroll
        for (int j = 0; j < 8; ++j) {
            float ig = fast_sigmoid(acc[0][j]);
            float fg = fast_sigmoid(acc[1][j]);
            float gg = fast_tanh   (acc[2][j]);
            float og = fast_sigmoid(acc[3][j]);
            float cn = fg * c[j] + ig * gg;
            c[j] = cn;
            float h = og * fast_tanh(cn);
            int m   = mlo + j;
            int col = hc * 16 + n;
            h_lds[m * HSTRIDE + col] = (bf16)h;
            HS[(size_t)(t * 16 + m) * HH + col] = (bf16)h;
        }
        __syncthreads();   // h(t) fully published before next step reads it
    }
}

// ------------- Phase 3: feats[b][t][nt] = h . W_out[nt] + b_out[nt] -------------
__global__ void feats_kernel(const bf16* __restrict__ HS,
                             const float* __restrict__ W_out,
                             const float* __restrict__ b_out,
                             float* __restrict__ feats) {
    int idx = blockIdx.x * blockDim.x + threadIdx.x;
    if (idx >= BB * TT * NTAG) return;
    int nt  = idx % NTAG;
    int row = idx / NTAG;               // t*16 + b
    int b = row & 15, t = row >> 4;
    const bf16*  h = HS + (size_t)row * HH;
    const float* w = W_out + (size_t)nt * HH;
    float s = b_out[nt];
#pragma unroll 8
    for (int k = 0; k < HH; ++k) s += (float)h[k] * w[k];
    feats[((size_t)b * TT + t) * NTAG + nt] = s;
}

// ------------- Phase 4: per-sequence Viterbi (NT=6), backpointers in LDS -------------
__global__ void viterbi_kernel(const float* __restrict__ feats,
                               const float* __restrict__ trans,
                               float* __restrict__ out) {
    __shared__ unsigned char bptr[TT][8];
    int b = blockIdx.x;
    if (threadIdx.x != 0) return;
    float tr[NTAG][NTAG];
    for (int i = 0; i < NTAG; ++i)
        for (int j = 0; j < NTAG; ++j) tr[i][j] = trans[i * NTAG + j];
    float fv[NTAG];
    for (int i = 0; i < NTAG; ++i) fv[i] = (i == START_TAG) ? 0.0f : NEGV;
    const float* F = feats + (size_t)b * TT * NTAG;
    for (int t = 0; t < TT; ++t) {
        float nfv[NTAG];
        for (int nx = 0; nx < NTAG; ++nx) {
            float best = fv[0] + tr[nx][0]; int bi = 0;
            for (int pv = 1; pv < NTAG; ++pv) {
                float s = fv[pv] + tr[nx][pv];
                if (s > best) { best = s; bi = pv; }
            }
            nfv[nx] = best + F[t * NTAG + nx];
            bptr[t][nx] = (unsigned char)bi;
        }
        for (int i = 0; i < NTAG; ++i) fv[i] = nfv[i];
    }
    float best = fv[0] + tr[STOP_TAG][0]; int bi = 0;
    for (int pv = 1; pv < NTAG; ++pv) {
        float s = fv[pv] + tr[STOP_TAG][pv];
        if (s > best) { best = s; bi = pv; }
    }
    out[b] = best;
    float* paths = out + BB;
    int tag = bi;
    for (int t = TT - 1; t >= 0; --t) {
        paths[(size_t)b * TT + t] = (float)tag;
        tag = bptr[t][tag];
    }
}

extern "C" void kernel_launch(void* const* d_in, const int* in_sizes, int n_in,
                              void* d_out, int out_size, void* d_ws, size_t ws_size,
                              hipStream_t stream) {
    const int*   sentence  = (const int*)d_in[0];
    const float* embedding = (const float*)d_in[1];
    const float* W_ih  = (const float*)d_in[2];
    const float* W_hh  = (const float*)d_in[3];
    const float* b_ih  = (const float*)d_in[4];
    const float* b_hh  = (const float*)d_in[5];
    const float* W_out = (const float*)d_in[6];
    const float* b_out = (const float*)d_in[7];
    const float* trans = (const float*)d_in[8];

    char* ws = (char*)d_ws;
    bf16*  E     = (bf16*)(ws);                                   // 4 MB (reused as HS)
    bf16*  HS    = E;
    bf16*  Wihb  = (bf16*)(ws + (4u  << 20));                     // 512 KB
    bf16*  Whhb  = (bf16*)(ws + (4u  << 20) + (512u << 10));      // 512 KB
    float* bias  = (float*)(ws + (5u  << 20));                    // 4 KB
    float* X     = (float*)(ws + (8u  << 20));                    // 32 MB
    float* feats = (float*)(ws + (40u << 20));                    // 192 KB
    float* out   = (float*)d_out;

    {
        int total = TT * BB * HH + 2 * G4H * HH + G4H;
        prep_kernel<<<(total + 255) / 256, 256, 0, stream>>>(
            sentence, embedding, W_ih, W_hh, b_ih, b_hh, E, Wihb, Whhb, bias);
    }
    gemm_x_kernel<<<4096, 256, 0, stream>>>(E, Wihb, bias, X);
    lstm_kernel<<<1, 512, 0, stream>>>(X, Whhb, HS);
    feats_kernel<<<(BB * TT * NTAG + 255) / 256, 256, 0, stream>>>(HS, W_out, b_out, feats);
    viterbi_kernel<<<BB, 32, 0, stream>>>(feats, trans, out);
}
